// AttentionBlock_10307921510604
// MI455X (gfx1250) — matrix operand order repaired and checked by reading, not compiled, not run on hardware
//
#include <hip/hip_runtime.h>
#include <cstdint>

typedef __attribute__((ext_vector_type(16))) __bf16 v16bf;
typedef __attribute__((ext_vector_type(8)))  float  v8f;

#define B_      4
#define T_      1024
#define DM      2048
#define NH      32
#define NKV     8
#define QKV_DIM 3072          // 64*(32+16)
#define WIN     128
#define SM_SCALE 0.125f       // 1/sqrt(64)

// ---------------- fp32 -> bf16 (round to nearest even) ----------------
__device__ __forceinline__ unsigned short f2bf(float x) {
    uint32_t u = __float_as_uint(x);
    uint32_t r = u + 0x7FFFu + ((u >> 16) & 1u);
    return (unsigned short)(r >> 16);
}

__global__ void cast_f32_to_bf16(const float* __restrict__ in,
                                 unsigned short* __restrict__ out, int n) {
    int i = blockIdx.x * blockDim.x + threadIdx.x;
    if (i < n) out[i] = f2bf(in[i]);
}

// ---------------- YaRN / NTK-by-parts rope tables ----------------
__global__ void rope_tables(float* __restrict__ cosT, float* __restrict__ sinT) {
    int tid = blockIdx.x * blockDim.x + threadIdx.x;
    if (tid >= T_ * 32) return;
    int t = tid >> 5, i = tid & 31;
    float di   = (float)i;
    float freq = powf(150000.0f, di * (1.0f / 32.0f));
    float conc = 0.1f * logf(32.0f) + 1.0f;
    const float PI2 = 6.28318530717958647692f;
    float lt   = logf(150000.0f);
    float low  = 32.0f * logf(1024.0f / (32.0f * PI2)) / lt;
    float high = 32.0f * logf(1024.0f / PI2) / lt;
    float ramp = fminf(fmaxf((di - low) / (high - low), 0.0f), 1.0f);
    float inv  = (1.0f / (32.0f * freq)) * ramp + (1.0f / freq) * (1.0f - ramp);
    float ang  = (float)t * inv;
    cosT[tid] = cosf(ang) * conc;
    sinT[tid] = sinf(ang) * conc;
}

// ---------------- in-place split-half RoPE on q (32 heads) + k (8 heads) ----
__global__ void rope_apply(float* __restrict__ qkv,
                           const float* __restrict__ cosT,
                           const float* __restrict__ sinT) {
    int tid = blockIdx.x * blockDim.x + threadIdx.x;
    const int total = B_ * T_ * 40 * 32;
    if (tid >= total) return;
    int i  = tid & 31;
    int hh = (tid >> 5) % 40;
    int bt = tid / (32 * 40);
    int t  = bt % T_;
    float* p = qkv + (size_t)bt * QKV_DIM + (hh < 32 ? hh * 64 : 2048 + (hh - 32) * 64);
    float c = cosT[t * 32 + i], s = sinT[t * 32 + i];
    float x1 = p[i], x2 = p[i + 32];
    p[i]      = x1 * c - x2 * s;
    p[i + 32] = x2 * c + x1 * s;
}

// ---------------- WMMA bf16 fragment loads (per ISA §7.12.2 layouts) -------
__device__ __forceinline__ v16bf ldfragA(const unsigned short* p) {
    // lane half 'lh' baked into p; elems 0..7 at +0, elems 8..15 at K+16
    union { uint4 q[2]; v16bf v; } u;
    u.q[0] = *reinterpret_cast<const uint4*>(p);
    u.q[1] = *reinterpret_cast<const uint4*>(p + 16);
    return u.v;
}
__device__ __forceinline__ v16bf ldfragB(const unsigned short* p) {
    // lane half 'lh' baked into p; elems 0..7 at +0, elems 8..15 at K+16
    union { uint4 q[2]; v16bf v; } u;
    u.q[0] = *reinterpret_cast<const uint4*>(p);
    u.q[1] = *reinterpret_cast<const uint4*>(p + 16);
    return u.v;
}

// C[M,N] = A[M,K](bf16) * W[N,K]^T(bf16) + bias[N], f32 out.
// Block: 256 thr = 8 waves, 128x128 tile; wave: 32x64 (2x4 fragments).
__global__ __launch_bounds__(256) void gemm_bf16(
    const unsigned short* __restrict__ A, const unsigned short* __restrict__ W,
    const float* __restrict__ bias, float* __restrict__ C, int M, int N, int K) {
    const int lane = threadIdx.x & 31;
    const int wave = threadIdx.x >> 5;
    const int wm = wave >> 1, wn = wave & 1;
    const int lr = lane & 15, lh = lane >> 4;
    const int m0 = blockIdx.y * 128 + wm * 32;
    const int n0 = blockIdx.x * 128 + wn * 64;

    v8f vzero = {0.f, 0.f, 0.f, 0.f, 0.f, 0.f, 0.f, 0.f};
    v8f acc[2][4];
#pragma unroll
    for (int a = 0; a < 2; ++a)
#pragma unroll
        for (int b = 0; b < 4; ++b) acc[a][b] = vzero;

    const unsigned short* ap0 = A + (size_t)(m0 + lr) * K + lh * 8;
    const unsigned short* ap1 = ap0 + (size_t)16 * K;
    const unsigned short* bp0 = W + (size_t)(n0 + lr) * K + lh * 8;
    const unsigned short* bp1 = bp0 + (size_t)16 * K;
    const unsigned short* bp2 = bp0 + (size_t)32 * K;
    const unsigned short* bp3 = bp0 + (size_t)48 * K;

    for (int k = 0; k < K; k += 32) {
        __builtin_prefetch(ap0 + k + 256, 0, 1);   // global_prefetch_b8
        __builtin_prefetch(bp0 + k + 256, 0, 1);
        v16bf a0 = ldfragA(ap0 + k);
        v16bf a1 = ldfragA(ap1 + k);
        v16bf b0 = ldfragB(bp0 + k);
        v16bf b1 = ldfragB(bp1 + k);
        v16bf b2 = ldfragB(bp2 + k);
        v16bf b3 = ldfragB(bp3 + k);
        acc[0][0] = __builtin_amdgcn_wmma_f32_16x16x32_bf16(false, a0, false, b0, (short)0, acc[0][0], false, false);
        acc[0][1] = __builtin_amdgcn_wmma_f32_16x16x32_bf16(false, a0, false, b1, (short)0, acc[0][1], false, false);
        acc[0][2] = __builtin_amdgcn_wmma_f32_16x16x32_bf16(false, a0, false, b2, (short)0, acc[0][2], false, false);
        acc[0][3] = __builtin_amdgcn_wmma_f32_16x16x32_bf16(false, a0, false, b3, (short)0, acc[0][3], false, false);
        acc[1][0] = __builtin_amdgcn_wmma_f32_16x16x32_bf16(false, a1, false, b0, (short)0, acc[1][0], false, false);
        acc[1][1] = __builtin_amdgcn_wmma_f32_16x16x32_bf16(false, a1, false, b1, (short)0, acc[1][1], false, false);
        acc[1][2] = __builtin_amdgcn_wmma_f32_16x16x32_bf16(false, a1, false, b2, (short)0, acc[1][2], false, false);
        acc[1][3] = __builtin_amdgcn_wmma_f32_16x16x32_bf16(false, a1, false, b3, (short)0, acc[1][3], false, false);
    }

#pragma unroll
    for (int mi = 0; mi < 2; ++mi) {
        int row = m0 + mi * 16 + lh * 8;
#pragma unroll
        for (int ni = 0; ni < 4; ++ni) {
            int col = n0 + ni * 16 + lr;
            float bv = bias[col];
#pragma unroll
            for (int r = 0; r < 8; ++r)
                C[(size_t)(row + r) * N + col] = acc[mi][ni][r] + bv;
        }
    }
}

// ---------------- sliding-window GQA attention with sinks ------------------
// one wave32 per (b, head, query); 8 queries per 256-thread block
__global__ __launch_bounds__(256) void attn_sliding(
    const float* __restrict__ qkv, const float* __restrict__ sinks,
    unsigned short* __restrict__ attn) {
    __shared__ float sw[8][128];
    const int lane = threadIdx.x & 31;
    const int wave = threadIdx.x >> 5;
    int bid = blockIdx.x;
    int tg = bid & 127;          // T_/8 = 128
    int h  = (bid >> 7) & 31;
    int b  = bid >> 12;
    int i   = tg * 8 + wave;
    int kvh = h >> 2;            // Q_MULT = 4

    const float* qp = qkv + ((size_t)(b * T_ + i)) * QKV_DIM + h * 64;
    float q[64];
#pragma unroll
    for (int d = 0; d < 64; ++d) q[d] = qp[d];

    int jlo = i - (WIN - 1); if (jlo < 0) jlo = 0;

    float lg[4];
#pragma unroll
    for (int c = 0; c < 4; ++c) {
        int j = jlo + c * 32 + lane;         // always < T_, memory-safe
        const float4* kp = (const float4*)(qkv + ((size_t)(b * T_ + j)) * QKV_DIM + 2048 + kvh * 64);
        float acc = 0.f;
#pragma unroll
        for (int d4 = 0; d4 < 16; ++d4) {
            float4 k4 = kp[d4];
            acc += q[4 * d4 + 0] * k4.x + q[4 * d4 + 1] * k4.y +
                   q[4 * d4 + 2] * k4.z + q[4 * d4 + 3] * k4.w;
        }
        lg[c] = (j <= i) ? acc * SM_SCALE : -__builtin_inff();
    }

    // wave32 max reduction
    float m = fmaxf(fmaxf(lg[0], lg[1]), fmaxf(lg[2], lg[3]));
#pragma unroll
    for (int off = 16; off > 0; off >>= 1) m = fmaxf(m, __shfl_xor(m, off, 32));

    float e[4], ssum = 0.f;
#pragma unroll
    for (int c = 0; c < 4; ++c) { e[c] = expf(lg[c] - m); ssum += e[c]; }
#pragma unroll
    for (int off = 16; off > 0; off >>= 1) ssum += __shfl_xor(ssum, off, 32);

    float denom = ssum + expf(sinks[h] - m);   // sink in denominator
    float inv = 1.0f / denom;
#pragma unroll
    for (int c = 0; c < 4; ++c) sw[wave][c * 32 + lane] = e[c] * inv;
    __syncthreads();

    int d0 = lane, d1 = lane + 32;
    float a0 = 0.f, a1 = 0.f;
    for (int jj = 0; jj < 128; ++jj) {
        float wgt = sw[wave][jj];              // uniform per wave
        if (wgt != 0.f) {
            const float* vp = qkv + ((size_t)(b * T_ + (jlo + jj))) * QKV_DIM + 2560 + kvh * 64;
            a0 += wgt * vp[d0];
            a1 += wgt * vp[d1];
        }
    }
    size_t orow = ((size_t)(b * T_ + i)) * DM + h * 64;
    attn[orow + d0] = f2bf(a0);
    attn[orow + d1] = f2bf(a1);
}

// ---------------------------------------------------------------------------
extern "C" void kernel_launch(void* const* d_in, const int* in_sizes, int n_in,
                              void* d_out, int out_size, void* d_ws, size_t ws_size,
                              hipStream_t stream) {
    (void)in_sizes; (void)n_in; (void)out_size; (void)ws_size;
    const float* x     = (const float*)d_in[0];
    const float* qkvw  = (const float*)d_in[1];
    const float* qkvb  = (const float*)d_in[2];
    const float* outw  = (const float*)d_in[3];
    const float* outb  = (const float*)d_in[4];
    const float* sinks = (const float*)d_in[5];
    float* out = (float*)d_out;

    const size_t M = (size_t)B_ * T_;   // 4096
    char* ws = (char*)d_ws;
    size_t off = 0;
    unsigned short* Xb = (unsigned short*)(ws + off); off += M * DM * 2;
    unsigned short* Wq = (unsigned short*)(ws + off); off += (size_t)QKV_DIM * DM * 2;
    unsigned short* Wo = (unsigned short*)(ws + off); off += (size_t)DM * DM * 2;
    float* QKV         = (float*)(ws + off);          off += M * QKV_DIM * 4;
    unsigned short* At = (unsigned short*)(ws + off); off += M * DM * 2;
    float* cosT        = (float*)(ws + off);          off += (size_t)T_ * 32 * 4;
    float* sinT        = (float*)(ws + off);          off += (size_t)T_ * 32 * 4;

    int n1 = (int)(M * DM);
    cast_f32_to_bf16<<<(n1 + 255) / 256, 256, 0, stream>>>(x, Xb, n1);
    int n2 = QKV_DIM * DM;
    cast_f32_to_bf16<<<(n2 + 255) / 256, 256, 0, stream>>>(qkvw, Wq, n2);
    int n3 = DM * DM;
    cast_f32_to_bf16<<<(n3 + 255) / 256, 256, 0, stream>>>(outw, Wo, n3);
    rope_tables<<<(T_ * 32 + 255) / 256, 256, 0, stream>>>(cosT, sinT);

    gemm_bf16<<<dim3(QKV_DIM / 128, (int)(M / 128)), 256, 0, stream>>>(
        Xb, Wq, qkvb, QKV, (int)M, QKV_DIM, DM);

    int n4 = B_ * T_ * 40 * 32;
    rope_apply<<<(n4 + 255) / 256, 256, 0, stream>>>(QKV, cosT, sinT);

    attn_sliding<<<B_ * NH * (T_ / 8), 256, 0, stream>>>(QKV, sinks, At);

    gemm_bf16<<<dim3(DM / 128, (int)(M / 128)), 256, 0, stream>>>(
        At, Wo, outb, out, (int)M, DM, DM);
}
